// SPNorm_20538533610198
// MI455X (gfx1250) — matrix operand order repaired and checked by reading, not compiled, not run on hardware
//
#include <hip/hip_runtime.h>
#include <hip/hip_bf16.h>

typedef __attribute__((ext_vector_type(16))) _Float16 v16h;
typedef __attribute__((ext_vector_type(8)))  _Float16 v8h;
typedef __attribute__((ext_vector_type(2)))  _Float16 h2;
typedef __attribute__((ext_vector_type(8)))  float    v8f;

#define C_CH       128
#define TILE_P     128
#define HW         65536            // 256*256 pixels per image
#define XS_STRIDE  132              // floats per channel row (128 + 4 pad)
#define N16_STRIDE 136              // halves per pixel row (128 + 8 pad), 272B = 16B aligned
#define XS_BYTES   (C_CH * XS_STRIDE * 4)          // 67584
#define N16_BYTES  (TILE_P * N16_STRIDE * 2)       // 34816
#define SMEM_BYTES (XS_BYTES + N16_BYTES)          // 102400 = 100KB -> 3 blocks/WGP

__global__ __launch_bounds__(128) void
spnorm_wmma_kernel(const float* __restrict__ x,
                   const float* __restrict__ W,
                   const float* __restrict__ bias,
                   float* __restrict__ out)
{
    extern __shared__ char smem[];
    float*    xs  = (float*)smem;                       // [C_CH][XS_STRIDE] fp32 x tile
    _Float16* n16 = (_Float16*)(smem + XS_BYTES);       // [TILE_P][N16_STRIDE] f16 normalized

    const int tid   = threadIdx.x;
    const int pix0  = blockIdx.x * TILE_P;              // flat pixel id (b*HW + hw)
    const int b_img = pix0 >> 16;
    const int hw0   = pix0 & 0xFFFF;
    const size_t base = (size_t)b_img * C_CH * HW + hw0;
    const float* xp = x   + base;                       // element [c][p] at xp[c*HW + p]
    float*       op = out + base;

    // ---- Phase 1: stream all channels of this thread's pixel (NT: single-use from HBM),
    //               accumulate stats, stash fp32 copy in LDS for phases 2 & 4.
    float sum = 0.f, sumsq = 0.f;
    #pragma unroll 8
    for (int c = 0; c < C_CH; ++c) {
        float v = __builtin_nontemporal_load(xp + (size_t)c * HW + tid);
        xs[c * XS_STRIDE + tid] = v;
        sum   += v;
        sumsq += v * v;
    }
    float mean = sum * (1.0f / 128.0f);
    float var  = (sumsq - sum * mean) * (1.0f / 127.0f);   // unbiased (ddof=1)
    var = fmaxf(var, 0.0f);
    float inv = 1.0f / (sqrtf(var) + 1e-6f);

    // ---- Phase 2: normalized f16, [pixel][channel] layout (B-fragment friendly)
    #pragma unroll 4
    for (int c = 0; c < C_CH; c += 2) {
        float n0 = (xs[(c    ) * XS_STRIDE + tid] - mean) * inv;
        float n1 = (xs[(c + 1) * XS_STRIDE + tid] - mean) * inv;
        h2 hh; hh.x = (_Float16)n0; hh.y = (_Float16)n1;
        *(h2*)(n16 + tid * N16_STRIDE + c) = hh;           // 4B packed store
    }
    __syncthreads();

    // ---- Phase 3: y = W * n via v_wmma_f32_16x16x32_f16
    //      4 waves, wave w owns M-tiles {2w, 2w+1} (out-channel rows 32w..32w+31)
    const int wave = tid >> 5;
    const int lane = tid & 31;
    const int ln15 = lane & 15;
    const int hi   = lane >> 4;                  // half-wave select

    // A fragments, documented 16-bit A layout:
    // lane<16: halves 0-7 -> K=kb..kb+7, halves 8-15 -> K=kb+16..kb+23 (kb=kk*32)
    // lane>=16: same +8
    v16h afrag[2][4];
    #pragma unroll
    for (int mt = 0; mt < 2; ++mt) {
        const int orow = (wave * 2 + mt) * 16 + ln15;      // W row this lane supplies
        const float4* wr = (const float4*)(W + orow * C_CH);
        #pragma unroll
        for (int kk = 0; kk < 4; ++kk) {
            const int kbase = kk * 32 + hi * 8;
            float4 q0 = wr[(kbase +  0) >> 2];
            float4 q1 = wr[(kbase +  4) >> 2];
            float4 q2 = wr[(kbase + 16) >> 2];
            float4 q3 = wr[(kbase + 20) >> 2];
            v16h a;
            a[0]  = (_Float16)q0.x; a[1]  = (_Float16)q0.y;
            a[2]  = (_Float16)q0.z; a[3]  = (_Float16)q0.w;
            a[4]  = (_Float16)q1.x; a[5]  = (_Float16)q1.y;
            a[6]  = (_Float16)q1.z; a[7]  = (_Float16)q1.w;
            a[8]  = (_Float16)q2.x; a[9]  = (_Float16)q2.y;
            a[10] = (_Float16)q2.z; a[11] = (_Float16)q2.w;
            a[12] = (_Float16)q3.x; a[13] = (_Float16)q3.y;
            a[14] = (_Float16)q3.z; a[15] = (_Float16)q3.w;
            afrag[mt][kk] = a;
        }
    }

    // Per-lane bias for the 8 accumulator rows of each M-tile (VGPR r -> M = r + 8*hi)
    float biasv[2][8];
    #pragma unroll
    for (int mt = 0; mt < 2; ++mt)
        #pragma unroll
        for (int r = 0; r < 8; ++r)
            biasv[mt][r] = bias[(wave * 2 + mt) * 16 + r + hi * 8];

    for (int pt = 0; pt < 8; ++pt) {             // 8 pixel tiles of 16
        const int p = pt * 16 + ln15;
        v8f acc0 = {};
        v8f acc1 = {};
        #pragma unroll
        for (int kk = 0; kk < 4; ++kk) {
            // B layout: half i<8 -> K = kk*32 + 8*hi + i, half i>=8 -> K = kk*32 + 16 + 8*hi + (i-8)
            const int cbase = kk * 32 + hi * 8;
            const _Float16* src = n16 + p * N16_STRIDE + cbase;
            v8h blo = *(const v8h*)(src);
            v8h bhi = *(const v8h*)(src + 16);
            v16h bfrag;
            #pragma unroll
            for (int i = 0; i < 8; ++i) { bfrag[i] = blo[i]; bfrag[i + 8] = bhi[i]; }
            acc0 = __builtin_amdgcn_wmma_f32_16x16x32_f16(
                false, afrag[0][kk], false, bfrag, (short)0, acc0, false, false);
            acc1 = __builtin_amdgcn_wmma_f32_16x16x32_f16(
                false, afrag[1][kk], false, bfrag, (short)0, acc1, false, false);
        }
        // ---- Phase 4: out = x * (y + b); x from LDS, NT stores (out never re-read)
        #pragma unroll
        for (int r = 0; r < 8; ++r) {
            const int o0 = (wave * 2 + 0) * 16 + r + hi * 8;
            const int o1 = (wave * 2 + 1) * 16 + r + hi * 8;
            float xv0 = xs[o0 * XS_STRIDE + p];
            float xv1 = xs[o1 * XS_STRIDE + p];
            __builtin_nontemporal_store(xv0 * (acc0[r] + biasv[0][r]),
                                        op + (size_t)o0 * HW + p);
            __builtin_nontemporal_store(xv1 * (acc1[r] + biasv[1][r]),
                                        op + (size_t)o1 * HW + p);
        }
    }
}

extern "C" void kernel_launch(void* const* d_in, const int* in_sizes, int n_in,
                              void* d_out, int out_size, void* d_ws, size_t ws_size,
                              hipStream_t stream)
{
    const float* x  = (const float*)d_in[0];   // [8,128,256,256] fp32
    const float* W  = (const float*)d_in[1];   // [128,128] fp32
    const float* b  = (const float*)d_in[2];   // [128] fp32
    float* out = (float*)d_out;

    const int n_pix  = in_sizes[0] / C_CH;     // 8*256*256 = 524288
    const int blocks = n_pix / TILE_P;         // 4096

    spnorm_wmma_kernel<<<blocks, 128, SMEM_BYTES, stream>>>(x, W, b, out);
}
